// FastMHA_20066087207297
// MI455X (gfx1250) — compile-verified, matrix operand order checked by reading, not run on hardware
//
#include <hip/hip_runtime.h>

typedef __attribute__((ext_vector_type(16))) __bf16 v16bf;
typedef __attribute__((ext_vector_type(8)))  __bf16 v8bf;
typedef __attribute__((ext_vector_type(4)))  __bf16 v4bf;
typedef __attribute__((ext_vector_type(8)))  float  v8f;

#define S_LEN   1024
#define D_MODEL 1024
#define NH      16
#define HD      64
#define NB      33
#define NBP     48                        // NB padded to 16-multiple for WMMA
#define SCALE   0.0721687836487032206f   // 1/sqrt(64*3)

// Fragment layout (CDNA5 ISA §7.12.2) for 16-bit A/B operands:
// elems 0..7 -> K = hi*8 + 0..7 ; elems 8..15 -> K = 16 + hi*8 + 0..7.
// A fragment = two contiguous 16-byte LDS reads per lane.
union frag_u { v16bf f; v8bf h[2]; };

__device__ __forceinline__ v16bf load_frag(const __bf16* rowp, int hi) {
    frag_u u;
    u.h[0] = *(const v8bf*)(rowp + hi * 8);
    u.h[1] = *(const v8bf*)(rowp + 16 + hi * 8);
    return u.f;
}

// ---------------------------------------------------------------------------
// Q/K/V projection: out_bf16[b,h,s,d] = (X @ W + bias), X f32 [2048,1024]
// block = 256 thr (8 waves), tile 128(M) x 64(N), K-step 32.
// ---------------------------------------------------------------------------
__launch_bounds__(256)
__global__ void proj_qkv_kernel(const float* __restrict__ X,
                                const float* __restrict__ W,
                                const float* __restrict__ bias,
                                __bf16* __restrict__ out) {
    __shared__ __bf16 As[128][40];    // [M][K]
    __shared__ __bf16 Bs_t[64][40];   // [N][K] transposed
    const int m0 = blockIdx.x * 128;
    const int n0 = blockIdx.y * 64;
    const int t = threadIdx.x;
    const int w = t >> 5, lane = t & 31, hi = lane >> 4, ln = lane & 15;
    v8f acc[4] = {};
    for (int k0 = 0; k0 < D_MODEL; k0 += 32) {
        if (k0 + 32 < D_MODEL) {  // WGP-scope prefetch of next K-step
            __builtin_prefetch(&X[(size_t)(m0 + (t >> 1)) * D_MODEL + k0 + 32 + (t & 1) * 16], 0, 3);
            __builtin_prefetch(&W[(size_t)(k0 + 32 + (t >> 6) * 8) * D_MODEL + n0 + (t & 63)], 0, 3);
        }
        __syncthreads();
        #pragma unroll
        for (int e = 0; e < 4; ++e) {           // A: 128x32 f32 -> bf16
            int idx = t + e * 256;
            int r = idx >> 3, c = (idx & 7) * 4;
            float4 xv = *(const float4*)&X[(size_t)(m0 + r) * D_MODEL + k0 + c];
            v4bf pk = {(__bf16)xv.x, (__bf16)xv.y, (__bf16)xv.z, (__bf16)xv.w};
            *(v4bf*)&As[r][c] = pk;
        }
        #pragma unroll
        for (int e = 0; e < 2; ++e) {           // B: 32x64, transposed into [N][K]
            int idx = t + e * 256;
            int c = idx & 63, r0 = (idx >> 6) * 4;
            v4bf pk;
            #pragma unroll
            for (int rr = 0; rr < 4; ++rr)
                pk[rr] = (__bf16)W[(size_t)(k0 + r0 + rr) * D_MODEL + n0 + c];
            *(v4bf*)&Bs_t[c][r0] = pk;
        }
        __syncthreads();
        v16bf a = load_frag(&As[w * 16 + ln][0], hi);
        #pragma unroll
        for (int nt = 0; nt < 4; ++nt) {
            v16bf bb = load_frag(&Bs_t[nt * 16 + ln][0], hi);
            acc[nt] = __builtin_amdgcn_wmma_f32_16x16x32_bf16(
                false, a, false, bb, (short)0, acc[nt], false, false);
        }
    }
    #pragma unroll
    for (int nt = 0; nt < 4; ++nt) {
        #pragma unroll
        for (int r = 0; r < 8; ++r) {
            int gm = m0 + w * 16 + r + hi * 8;
            int gn = n0 + nt * 16 + ln;
            float val = acc[nt][r] + bias[gn];
            int b = gm >> 10, s = gm & 1023;
            int h = gn >> 6,  dd = gn & 63;
            out[(((size_t)(b * NH + h)) * S_LEN + s) * HD + dd] = (__bf16)val;
        }
    }
}

// ---------------------------------------------------------------------------
// Output projection: out_f32[2048,1024] = ctx_bf16 @ Wo + bo
// ---------------------------------------------------------------------------
__launch_bounds__(256)
__global__ void proj_out_kernel(const __bf16* __restrict__ X,
                                const float* __restrict__ W,
                                const float* __restrict__ bias,
                                float* __restrict__ out) {
    __shared__ __bf16 As[128][40];
    __shared__ __bf16 Bs_t[64][40];
    const int m0 = blockIdx.x * 128;
    const int n0 = blockIdx.y * 64;
    const int t = threadIdx.x;
    const int w = t >> 5, lane = t & 31, hi = lane >> 4, ln = lane & 15;
    v8f acc[4] = {};
    for (int k0 = 0; k0 < D_MODEL; k0 += 32) {
        if (k0 + 32 < D_MODEL) {
            __builtin_prefetch(&X[(size_t)(m0 + (t >> 1)) * D_MODEL + k0 + 32], 0, 3);
            __builtin_prefetch(&W[(size_t)(k0 + 32 + (t >> 6) * 8) * D_MODEL + n0 + (t & 63)], 0, 3);
        }
        __syncthreads();
        #pragma unroll
        for (int e = 0; e < 2; ++e) {           // A: bf16, 128x32
            int idx = t + e * 256;
            int r = idx >> 2, c = (idx & 3) * 8;
            *(v8bf*)&As[r][c] = *(const v8bf*)&X[(size_t)(m0 + r) * D_MODEL + k0 + c];
        }
        #pragma unroll
        for (int e = 0; e < 2; ++e) {
            int idx = t + e * 256;
            int c = idx & 63, r0 = (idx >> 6) * 4;
            v4bf pk;
            #pragma unroll
            for (int rr = 0; rr < 4; ++rr)
                pk[rr] = (__bf16)W[(size_t)(k0 + r0 + rr) * D_MODEL + n0 + c];
            *(v4bf*)&Bs_t[c][r0] = pk;
        }
        __syncthreads();
        v16bf a = load_frag(&As[w * 16 + ln][0], hi);
        #pragma unroll
        for (int nt = 0; nt < 4; ++nt) {
            v16bf bb = load_frag(&Bs_t[nt * 16 + ln][0], hi);
            acc[nt] = __builtin_amdgcn_wmma_f32_16x16x32_bf16(
                false, a, false, bb, (short)0, acc[nt], false, false);
        }
    }
    #pragma unroll
    for (int nt = 0; nt < 4; ++nt) {
        #pragma unroll
        for (int r = 0; r < 8; ++r) {
            int gm = m0 + w * 16 + r + hi * 8;
            int gn = n0 + nt * 16 + ln;
            out[(size_t)gm * D_MODEL + gn] = acc[nt][r] + bias[gn];
        }
    }
}

// ---------------------------------------------------------------------------
// Relative-position raw tables via WMMA (buckets padded 33->48):
//   raw_c2p[bh,i,k] = scale * (Q  @ rel_k^T)      [64 x 48] per tile
//   raw_p2c[bh,k,j] = scale * (rel_q @ K^T)       [48 x 64] per tile
// grid (B*H, S/64), block 128 (4 waves). rel tables row-major in LDS are
// simultaneously the B-operand (N-major) layout for c2p and A layout for p2c.
// ---------------------------------------------------------------------------
__launch_bounds__(128)
__global__ void rel_wmma_kernel(const __bf16* __restrict__ Qbf,
                                const __bf16* __restrict__ Kbf,
                                const float* __restrict__ rel_q,
                                const float* __restrict__ rel_k,
                                float* __restrict__ raw_c2p,
                                float* __restrict__ raw_p2c) {
    __shared__ __bf16 Qs[64][72];
    __shared__ __bf16 Ks[64][72];
    __shared__ __bf16 RqS[NBP][72];
    __shared__ __bf16 RkS[NBP][72];
    const int bh = blockIdx.x;
    const int h = bh & (NH - 1);
    const int i0 = blockIdx.y * 64;
    const int t = threadIdx.x;
    const int w = t >> 5, lane = t & 31, hi = lane >> 4, ln = lane & 15;

    #pragma unroll
    for (int e = 0; e < 4; ++e) {                 // Q,K tiles 64x64
        int idx = t + e * 128;
        int r = idx >> 3, c = (idx & 7) * 8;
        *(v8bf*)&Qs[r][c] = *(const v8bf*)&Qbf[((size_t)bh * S_LEN + i0 + r) * HD + c];
        *(v8bf*)&Ks[r][c] = *(const v8bf*)&Kbf[((size_t)bh * S_LEN + i0 + r) * HD + c];
    }
    #pragma unroll
    for (int e = 0; e < 3; ++e) {                 // rel tables 48x64 (zero-padded)
        int idx = t + e * 128;
        int r = idx >> 3, g = (idx & 7) * 8;
        v8bf p1, p2;
        #pragma unroll
        for (int jj = 0; jj < 8; ++jj) {
            p1[jj] = (r < NB) ? (__bf16)rel_q[((size_t)h * NB + r) * HD + g + jj] : (__bf16)0.0f;
            p2[jj] = (r < NB) ? (__bf16)rel_k[((size_t)h * NB + r) * HD + g + jj] : (__bf16)0.0f;
        }
        *(v8bf*)&RqS[r][g] = p1;
        *(v8bf*)&RkS[r][g] = p2;
    }
    __syncthreads();

    // ---- c2p: rows = q rows of this wave, cols = buckets ----
    v8f cacc[3] = {};
    #pragma unroll
    for (int ks = 0; ks < HD; ks += 32) {
        v16bf a = load_frag(&Qs[w * 16 + ln][ks], hi);
        #pragma unroll
        for (int nt = 0; nt < 3; ++nt) {
            v16bf bb = load_frag(&RkS[nt * 16 + ln][ks], hi);
            cacc[nt] = __builtin_amdgcn_wmma_f32_16x16x32_bf16(
                false, a, false, bb, (short)0, cacc[nt], false, false);
        }
    }
    #pragma unroll
    for (int nt = 0; nt < 3; ++nt) {
        #pragma unroll
        for (int r = 0; r < 8; ++r) {
            int k = nt * 16 + ln;
            int i = i0 + w * 16 + r + hi * 8;
            if (k < NB)
                raw_c2p[((size_t)bh * S_LEN + i) * NB + k] = cacc[nt][r] * SCALE;
        }
    }

    // ---- p2c: rows = buckets, cols = k rows of this wave ----
    v8f pacc[3] = {};
    #pragma unroll
    for (int ks = 0; ks < HD; ks += 32) {
        v16bf bb = load_frag(&Ks[w * 16 + ln][ks], hi);
        #pragma unroll
        for (int mt = 0; mt < 3; ++mt) {
            v16bf a = load_frag(&RqS[mt * 16 + ln][ks], hi);
            pacc[mt] = __builtin_amdgcn_wmma_f32_16x16x32_bf16(
                false, a, false, bb, (short)0, pacc[mt], false, false);
        }
    }
    #pragma unroll
    for (int mt = 0; mt < 3; ++mt) {
        #pragma unroll
        for (int r = 0; r < 8; ++r) {
            int k = mt * 16 + r + hi * 8;
            int j = i0 + w * 16 + ln;
            if (k < NB)
                raw_p2c[((size_t)bh * NB + k) * S_LEN + j] = pacc[mt][r] * SCALE;
        }
    }
}

// ---------------------------------------------------------------------------
// Fused flash attention with disentangled relative terms.
// grid (B*H, S/64), block 128 (4 waves); wave w owns rows [w*16, w*16+16).
// ---------------------------------------------------------------------------
__launch_bounds__(128)
__global__ void attn_kernel(const __bf16* __restrict__ Qbf,
                            const __bf16* __restrict__ Kbf,
                            const __bf16* __restrict__ Vbf,
                            const float* __restrict__ raw_c2p,
                            const float* __restrict__ raw_p2c,
                            const int*   __restrict__ pos,
                            const unsigned char* __restrict__ mask,
                            const float* __restrict__ rel_v,
                            __bf16* __restrict__ ctx) {
    __shared__ __bf16 Qs[64][72];       // q rows; reused as rel_v^T in epilogue
    __shared__ __bf16 Ks[64][72];       // k rows (N-major for B operand)
    __shared__ __bf16 Vs_t[64][72];     // v transposed: [dd][j]
    __shared__ __bf16 Ps[4][16][72];    // per-wave P staging; reused as agg_bf16
    __shared__ float  agg[64][34];      // per-row bucket accumulator
    const int bh = blockIdx.x;
    const int b = bh >> 4, h = bh & 15;
    const int i0 = blockIdx.y * 64;
    const int t = threadIdx.x;
    const int w = t >> 5, lane = t & 31, hi = lane >> 4, ln = lane & 15;

    #pragma unroll
    for (int e = 0; e < 4; ++e) {
        int idx = t + e * 128;
        int r = idx >> 3, c = (idx & 7) * 8;
        *(v8bf*)&Qs[r][c] = *(const v8bf*)&Qbf[((size_t)bh * S_LEN + i0 + r) * HD + c];
    }
    for (int idx = t; idx < 64 * 34; idx += 128) ((float*)agg)[idx] = 0.f;

    v8f Oacc[4] = {};
    float m_run[8], l_run[8];
    #pragma unroll
    for (int r = 0; r < 8; ++r) { m_run[r] = -1e30f; l_run[r] = 0.f; }
    __syncthreads();

    for (int j0 = 0; j0 < S_LEN; j0 += 64) {
        if (j0 + 64 < S_LEN)
            __builtin_prefetch(&Kbf[((size_t)bh * S_LEN + j0 + 64 + (t >> 1)) * HD], 0, 3);
        __syncthreads();
        #pragma unroll
        for (int e = 0; e < 4; ++e) {                // K rows (row-major)
            int idx = t + e * 128;
            int r = idx >> 3, c = (idx & 7) * 8;
            *(v8bf*)&Ks[r][c] = *(const v8bf*)&Kbf[((size_t)bh * S_LEN + j0 + r) * HD + c];
        }
        {                                            // V transposed: Vs_t[dd][j]
            int dd = t & 63, half = t >> 6;
            #pragma unroll
            for (int e = 0; e < 4; ++e) {
                int jb = half * 32 + e * 8;
                v8bf pk;
                #pragma unroll
                for (int jj = 0; jj < 8; ++jj)
                    pk[jj] = Vbf[((size_t)bh * S_LEN + j0 + jb + jj) * HD + dd];
                *(v8bf*)&Vs_t[dd][jb] = pk;
            }
        }
        __syncthreads();

        // ---- scores = q @ k^T ----
        v8f sacc[4] = {};
        #pragma unroll
        for (int ks = 0; ks < HD; ks += 32) {
            v16bf a = load_frag(&Qs[w * 16 + ln][ks], hi);
            #pragma unroll
            for (int nt = 0; nt < 4; ++nt) {
                v16bf bb = load_frag(&Ks[nt * 16 + ln][ks], hi);
                sacc[nt] = __builtin_amdgcn_wmma_f32_16x16x32_bf16(
                    false, a, false, bb, (short)0, sacc[nt], false, false);
            }
        }

        // ---- gathered rel terms + mask ----
        float s_t[4][8]; int id_t[4][8];
        #pragma unroll
        for (int nt = 0; nt < 4; ++nt) {
            #pragma unroll
            for (int r = 0; r < 8; ++r) {
                int grow = i0 + w * 16 + r + hi * 8;
                int gj = j0 + nt * 16 + ln;
                int idx = pos[((size_t)b * S_LEN + grow) * S_LEN + gj];
                id_t[nt][r] = idx;
                float sc = sacc[nt][r] * SCALE
                         + raw_p2c[((size_t)bh * NB + idx) * S_LEN + gj]
                         + raw_c2p[((size_t)bh * S_LEN + grow) * NB + idx];
                if (mask[b * S_LEN + gj]) sc = -1e9f;
                s_t[nt][r] = sc;
            }
        }

        // ---- online softmax (row group = 16-lane half-wave) ----
        #pragma unroll
        for (int r = 0; r < 8; ++r) {
            float mloc = fmaxf(fmaxf(s_t[0][r], s_t[1][r]), fmaxf(s_t[2][r], s_t[3][r]));
            #pragma unroll
            for (int off = 1; off < 16; off <<= 1)
                mloc = fmaxf(mloc, __shfl_xor(mloc, off, 32));
            float m_new = fmaxf(m_run[r], mloc);
            float f = __expf(m_run[r] - m_new);
            float psum = 0.f;
            #pragma unroll
            for (int nt = 0; nt < 4; ++nt) {
                float p = __expf(s_t[nt][r] - m_new);
                s_t[nt][r] = p;
                psum += p;
            }
            #pragma unroll
            for (int off = 1; off < 16; off <<= 1)
                psum += __shfl_xor(psum, off, 32);
            l_run[r] = l_run[r] * f + psum;
            m_run[r] = m_new;
            #pragma unroll
            for (int nt = 0; nt < 4; ++nt) Oacc[nt][r] *= f;
            int wrow = w * 16 + r + hi * 8;
            for (int kq = ln; kq < NB; kq += 16) agg[wrow][kq] *= f;
        }

        // ---- scatter P into buckets + stage P in A-layout ----
        #pragma unroll
        for (int nt = 0; nt < 4; ++nt) {
            #pragma unroll
            for (int r = 0; r < 8; ++r) {
                float p = s_t[nt][r];
                int row16 = r + hi * 8;
                atomicAdd(&agg[w * 16 + row16][id_t[nt][r]], p);
                Ps[w][row16][nt * 16 + ln] = (__bf16)p;
            }
        }

        // ---- O += P @ V ----
        #pragma unroll
        for (int ks = 0; ks < 64; ks += 32) {
            v16bf a = load_frag(&Ps[w][ln][ks], hi);
            #pragma unroll
            for (int nt = 0; nt < 4; ++nt) {
                v16bf bb = load_frag(&Vs_t[nt * 16 + ln][ks], hi);
                Oacc[nt] = __builtin_amdgcn_wmma_f32_16x16x32_bf16(
                    false, a, false, bb, (short)0, Oacc[nt], false, false);
            }
        }
    }

    // ================= epilogue: O += agg @ rel_v (as WMMA) =================
    __syncthreads();
    {                  // stage rel_v^T (zero-padded K 33->64) into Qs buffer
        __bf16 (*Rv_t)[72] = Qs;                    // [dd][k]
        int dd = t & 63, half = t >> 6;
        #pragma unroll
        for (int e = 0; e < 4; ++e) {
            int kb = half * 32 + e * 8;
            v8bf pk;
            #pragma unroll
            for (int jj = 0; jj < 8; ++jj) {
                int kq = kb + jj;
                pk[jj] = (kq < NB) ? (__bf16)rel_v[((size_t)h * NB + kq) * HD + dd]
                                   : (__bf16)0.0f;
            }
            *(v8bf*)&Rv_t[dd][kb] = pk;
        }
    }
    {                  // convert this wave's agg rows to bf16 into Ps[w]
        #pragma unroll
        for (int e = 0; e < 4; ++e) {
            int li = e * 32 + lane;
            int row16 = li >> 3, g = (li & 7) * 8;
            v8bf pk;
            #pragma unroll
            for (int jj = 0; jj < 8; ++jj) {
                int kq = g + jj;
                pk[jj] = (kq < NB) ? (__bf16)agg[w * 16 + row16][kq] : (__bf16)0.0f;
            }
            *(v8bf*)&Ps[w][row16][g] = pk;
        }
    }
    __syncthreads();
    __bf16 (*Rv_t)[72] = Qs;
    #pragma unroll
    for (int ks = 0; ks < 64; ks += 32) {
        v16bf a = load_frag(&Ps[w][ln][ks], hi);
        #pragma unroll
        for (int nt = 0; nt < 4; ++nt) {
            v16bf bb = load_frag(&Rv_t[nt * 16 + ln][ks], hi);
            Oacc[nt] = __builtin_amdgcn_wmma_f32_16x16x32_bf16(
                false, a, false, bb, (short)0, Oacc[nt], false, false);
        }
    }

    // ---- normalize + write ctx ----
    #pragma unroll
    for (int nt = 0; nt < 4; ++nt) {
        #pragma unroll
        for (int r = 0; r < 8; ++r) {
            int row16 = r + hi * 8;
            int grow = i0 + w * 16 + row16;
            int dd = nt * 16 + ln;
            float val = Oacc[nt][r] / l_run[r];
            ctx[((size_t)b * S_LEN + grow) * D_MODEL + h * HD + dd] = (__bf16)val;
        }
    }
}

// ---------------------------------------------------------------------------
extern "C" void kernel_launch(void* const* d_in, const int* in_sizes, int n_in,
                              void* d_out, int out_size, void* d_ws, size_t ws_size,
                              hipStream_t stream) {
    const float* hidden = (const float*)d_in[0];
    const int*   pos    = (const int*)d_in[1];
    const unsigned char* mask = (const unsigned char*)d_in[2];
    const float* Wq = (const float*)d_in[3];
    const float* bq = (const float*)d_in[4];
    const float* Wk = (const float*)d_in[5];
    const float* bk = (const float*)d_in[6];
    const float* Wv = (const float*)d_in[7];
    const float* bv = (const float*)d_in[8];
    const float* Wo = (const float*)d_in[9];
    const float* bo = (const float*)d_in[10];
    const float* relq = (const float*)d_in[11];
    const float* relk = (const float*)d_in[12];
    const float* relv = (const float*)d_in[13];

    char* ws = (char*)d_ws;
    const size_t MB = 1048576;
    __bf16* Qbf = (__bf16*)(ws);                         // 4 MB
    __bf16* Kbf = (__bf16*)(ws + 4 * MB);                // 4 MB
    __bf16* Vbf = (__bf16*)(ws + 8 * MB);                // 4 MB
    float*  c2p = (float*)(ws + 12 * MB);                // 4.125 MB
    float*  p2c = (float*)(ws + 12 * MB + 4325376);      // 4.125 MB
    __bf16* ctx = (__bf16*)(ws + 12 * MB + 2 * 4325376); // 4 MB

    dim3 gProj(16, 16);   // M=2048/128, N=1024/64
    proj_qkv_kernel<<<gProj, 256, 0, stream>>>(hidden, Wq, bq, Qbf);
    proj_qkv_kernel<<<gProj, 256, 0, stream>>>(hidden, Wk, bk, Kbf);
    proj_qkv_kernel<<<gProj, 256, 0, stream>>>(hidden, Wv, bv, Vbf);
    rel_wmma_kernel<<<dim3(32, 16), 128, 0, stream>>>(Qbf, Kbf, relq, relk, c2p, p2c);
    attn_kernel<<<dim3(32, 16), 128, 0, stream>>>(Qbf, Kbf, Vbf, c2p, p2c, pos, mask,
                                                  relv, ctx);
    proj_out_kernel<<<gProj, 256, 0, stream>>>(ctx, Wo, bo, (float*)d_out);
}
